// Decoder_5299989643488
// MI455X (gfx1250) — matrix operand order repaired and checked by reading, not compiled, not run on hardware
//
#include <hip/hip_runtime.h>
#include <hip/hip_bf16.h>
#include <math.h>
#include <stdint.h>

typedef __attribute__((ext_vector_type(16))) _Float16 v16h;
typedef __attribute__((ext_vector_type(8)))  _Float16 v8h;
typedef __attribute__((ext_vector_type(8)))  float    v8f;
typedef __attribute__((ext_vector_type(4)))  unsigned int u32x4;
typedef __attribute__((ext_vector_type(8)))  int i32x8;
typedef __attribute__((ext_vector_type(4)))  int i32x4;

#define GB 64
#define GN 100
#define GD 32
#define ROWS (GB * GN)   // 6400
#define EKPAD 96         // 65 padded to 3 K-chunks of 32 (col 65 = bias row)
#define EHID 64
#define NKPAD 96
#define NHID 64
#define NOUT 32
#define LALPHA 0.1f

#if defined(__HIP_DEVICE_COMPILE__) && \
    __has_builtin(__builtin_amdgcn_tensor_load_to_lds) && \
    __has_builtin(__builtin_amdgcn_s_wait_tensorcnt)
#define HAVE_TDM 1
#else
#define HAVE_TDM 0
#endif

union AFrag { v16h v; v8h h[2]; };

__device__ __forceinline__ v8f wmma16(v16h a, v16h b, v8f c) {
  return __builtin_amdgcn_wmma_f32_16x16x32_f16(false, a, false, b, (short)0, c,
                                                false, false);
}

// leaky_relu(x) = max(x, alpha*x) for 0 < alpha < 1
__device__ __forceinline__ v8f leaky8(v8f x) {
  v8f r;
#pragma unroll
  for (int i = 0; i < 8; ++i) r[i] = fmaxf(x[i], LALPHA * x[i]);
  return r;
}

// argument is always a normal positive float (>= 1e-12): skip the precise
// sqrt's denorm-scaling + class-check fixup, use raw v_sqrt_f32.
__device__ __forceinline__ float fast_sqrtf(float x) {
#if defined(__HIP_DEVICE_COMPILE__) && __has_builtin(__builtin_amdgcn_sqrtf)
  return __builtin_amdgcn_sqrtf(x);
#else
  return sqrtf(x);
#endif
}

// A fragment gather, contiguous K order (matches B loads):
// lanes 0-15 : K = k0+{0..15}
// lanes 16-31: K = k0+{16..31}
__device__ __forceinline__ v16h lds_a_frag(const _Float16* rowk0, int hsel) {
  AFrag u;
  u.h[0] = *(const v8h*)(rowk0 + hsel * 16);
  u.h[1] = *(const v8h*)(rowk0 + hsel * 16 + 8);
  return u.v;
}

#if HAVE_TDM
// TDM: async DMA of a [rows<=16] x 32 f32 tile from global h into LDS.
// group0 = {count|flags, lds_addr, gaddr_lo, gaddr_hi|type=2}; group1 packs
// data_size=4B, tensor_dim0=32, tensor_dim1=rows remaining (OOB rows read as
// zero), tile = 32 x 16, dim0 stride = 32.
__device__ __forceinline__ void tdm_load_tile(const float* gsrc, unsigned lds_off,
                                              unsigned rows_remaining) {
  unsigned long long ga = (unsigned long long)(uintptr_t)gsrc;
  u32x4 g0;
  g0[0] = 1u;
  g0[1] = lds_off;
  g0[2] = (unsigned)(ga & 0xffffffffu);
  g0[3] = (unsigned)((ga >> 32) & 0x01ffffffu) | (2u << 30);
  i32x8 g1;
  g1[0] = (int)(2u << 16);                              // data_size = 4 bytes
  g1[1] = (int)((unsigned)GD << 16);                    // tensor_dim0 = 32
  g1[2] = (int)(rows_remaining << 16);                  // tensor_dim1 lo16
  g1[3] = (int)(((rows_remaining >> 16) & 0xffffu) | ((unsigned)GD << 16));
  g1[4] = 16;                                           // tile_dim1 = 16
  g1[5] = GD;                                           // tensor_dim0_stride
  g1[6] = 0;
  g1[7] = 0;
  i32x4 z4 = {0, 0, 0, 0};
#if __clang_major__ >= 23
  i32x8 z8 = {0, 0, 0, 0, 0, 0, 0, 0};
  __builtin_amdgcn_tensor_load_to_lds(g0, g1, z4, z4, z8, 0);
#else
  __builtin_amdgcn_tensor_load_to_lds(g0, g1, z4, z4, 0);
#endif
}
#endif

// ---------------------------------------------------------------------------
__global__ void wt_convert_kernel(const float* __restrict__ w,
                                  const float* __restrict__ bias,
                                  _Float16* __restrict__ wt,
                                  int K, int N, int Kpad, int biasRow) {
  int idx = blockIdx.x * blockDim.x + threadIdx.x;
  if (idx >= N * Kpad) return;
  int n = idx / Kpad, k = idx % Kpad;
  float v = 0.0f;
  if (k < K) v = w[k * N + n];
  else if (k == biasRow) v = bias[n];
  wt[idx] = (_Float16)v;
}

__global__ void lin_kernel(const float* __restrict__ x,
                           const float* __restrict__ w,
                           const float* __restrict__ bias,
                           float* __restrict__ h) {
  int idx = blockIdx.x * blockDim.x + threadIdx.x;
  if (idx >= GB * GN * GD) return;
  int b = idx / (GN * GD), c = idx % (GN * GD);
  const float* xr = x + b * GD;
  float s = bias[c];
#pragma unroll
  for (int k = 0; k < GD; ++k) s += xr[k] * w[k * (GN * GD) + c];
  h[idx] = s;
}

__global__ void out_kernel(const float* __restrict__ h,
                           const float* __restrict__ w,
                           const float* __restrict__ bias,
                           float* __restrict__ out) {
  int idx = blockIdx.x * blockDim.x + threadIdx.x;
  if (idx >= ROWS * 3) return;
  int row = idx / 3, o = idx % 3;
  const float* hr = h + (size_t)row * NOUT;
  float s = bias[o];
#pragma unroll
  for (int k = 0; k < NOUT; ++k) s += hr[k] * w[k * 3 + o];
  out[idx] = tanhf(s);
}

// ---------------------------------------------------------------------------
// Edge MLP + sum over j; one wave per (b, i); two j-tiles per iteration
// (8 independent WMMA accumulator chains per layer to fill hazard slots).
// x_j tiles are TDM-prefetched 2 ahead into 4 LDS buffers.
// ---------------------------------------------------------------------------
__global__ __launch_bounds__(32) void edge_mp_kernel(
    const float* __restrict__ h,        // [ROWS][32]
    const _Float16* __restrict__ w0t,   // [64][EKPAD], bias folded at K=65
    const _Float16* __restrict__ w1t,   // [64][64]
    const float* __restrict__ b1,       // [64]
    float* __restrict__ agg)            // [ROWS][64]
{
#if HAVE_TDM
  __shared__ __align__(16) float    hjbuf[4][16][GD];
#endif
  __shared__ __align__(16) _Float16 ein[2][16][EKPAD];
  __shared__ __align__(16) _Float16 act0[2][16][EHID];

  const int lane = threadIdx.x;
  const int r    = lane & 15;
  const int hsel = lane >> 4;
  const int ks   = hsel * 16;
  const int bi   = blockIdx.x;           // b*100 + i
  const int bb   = bi / GN;
  const float* __restrict__ hrow  = h + (size_t)bi * GD;
  const float* __restrict__ hbase = h + (size_t)bb * GN * GD;

  // x_i half-row in registers (loop invariant)
  float hik[16];
  {
    const float4* hp = (const float4*)(hrow + ks);
#pragma unroll
    for (int q = 0; q < 4; ++q) {
      float4 v = hp[q];
      hik[4 * q + 0] = v.x; hik[4 * q + 1] = v.y;
      hik[4 * q + 2] = v.z; hik[4 * q + 3] = v.w;
    }
  }
  // invariant e_in columns in both buffers: x_i, bias-one col 65, zero pad
  {
    v8h p0, p1;
#pragma unroll
    for (int e = 0; e < 8; ++e) {
      p0[e] = (_Float16)hik[e];
      p1[e] = (_Float16)hik[8 + e];
    }
#pragma unroll
    for (int q = 0; q < 2; ++q) {
      *(v8h*)&ein[q][r][ks]     = p0;
      *(v8h*)&ein[q][r][ks + 8] = p1;
    }
  }
  for (int idx = lane; idx < 2 * 16 * 31; idx += 32) {
    int q = idx / (16 * 31), rest = idx % (16 * 31);
    int rr = rest / 31, c = 65 + rest % 31;
    ein[q][rr][c] = (c == 65) ? (_Float16)1.0f : (_Float16)0.0f;
  }

  // weight fragments, reused across all j-tiles
  v16h w0f[3][4];
#pragma unroll
  for (int kk = 0; kk < 3; ++kk)
#pragma unroll
    for (int nt = 0; nt < 4; ++nt)
      w0f[kk][nt] = *(const v16h*)(w0t + (size_t)(nt * 16 + r) * EKPAD + kk * 32 + hsel * 16);
  v16h w1f[2][4];
#pragma unroll
  for (int kk = 0; kk < 2; ++kk)
#pragma unroll
    for (int nt = 0; nt < 4; ++nt)
      w1f[kk][nt] = *(const v16h*)(w1t + (size_t)(nt * 16 + r) * EHID + kk * 32 + hsel * 16);

  float bia1[4];
#pragma unroll
  for (int nt = 0; nt < 4; ++nt) bia1[nt] = b1[nt * 16 + r];

  v8f aggc[4];
  const v8f vzero = {};
#pragma unroll
  for (int nt = 0; nt < 4; ++nt) aggc[nt] = vzero;

#if HAVE_TDM
  const unsigned ldsb[4] = {(unsigned)(uintptr_t)&hjbuf[0][0][0],
                            (unsigned)(uintptr_t)&hjbuf[1][0][0],
                            (unsigned)(uintptr_t)&hjbuf[2][0][0],
                            (unsigned)(uintptr_t)&hjbuf[3][0][0]};
  tdm_load_tile(hbase,           ldsb[0], (unsigned)(ROWS - bb * GN));
  tdm_load_tile(hbase + 16 * GD, ldsb[1], (unsigned)(ROWS - bb * GN - 16));
#endif
  __syncthreads();

  // --- 3 full tile-pairs: j = 0..95 (all rows valid) -----------------------
  for (int p = 0; p < 3; ++p) {
    const int t0 = 2 * p;
#if HAVE_TDM
    if (p < 2) {            // prefetch two tiles ahead
      const int tn = t0 + 2;
      tdm_load_tile(hbase + (size_t)(tn * 16) * GD, ldsb[tn & 3],
                    (unsigned)(ROWS - (bb * GN + tn * 16)));
      tdm_load_tile(hbase + (size_t)(tn * 16 + 16) * GD, ldsb[(tn + 1) & 3],
                    (unsigned)(ROWS - (bb * GN + tn * 16 + 16)));
      __builtin_amdgcn_s_wait_tensorcnt(2);
    } else {                // prefetch the tail tile (t=6)
      tdm_load_tile(hbase + (size_t)(6 * 16) * GD, ldsb[2],
                    (unsigned)(ROWS - (bb * GN + 96)));
      __builtin_amdgcn_s_wait_tensorcnt(1);
    }
    __syncthreads();
#endif
    // build both e_in tiles: x_j columns + dist
#pragma unroll
    for (int q = 0; q < 2; ++q) {
      const int tt = t0 + q;
      float hj[16];
#if HAVE_TDM
      const float4* hjp = (const float4*)&hjbuf[tt & 3][r][ks];
#else
      const float* hjr = hbase + (size_t)(tt * 16 + r) * GD;  // j < 96 here
      const float4* hjp = (const float4*)(hjr + ks);
#endif
#pragma unroll
      for (int qq = 0; qq < 4; ++qq) {
        float4 v = hjp[qq];
        hj[4 * qq + 0] = v.x; hj[4 * qq + 1] = v.y;
        hj[4 * qq + 2] = v.z; hj[4 * qq + 3] = v.w;
      }
      float ss = 0.0f;
#pragma unroll
      for (int e = 0; e < 16; ++e) {
        float d = hik[e] - hj[e];
        ss += d * d;
      }
      v8h q0, q1;
#pragma unroll
      for (int e = 0; e < 8; ++e) {
        q0[e] = (_Float16)hj[e];
        q1[e] = (_Float16)hj[8 + e];
      }
      *(v8h*)&ein[q][r][GD + ks]     = q0;
      *(v8h*)&ein[q][r][GD + ks + 8] = q1;
      float so = __shfl_xor(ss, 16, 32);
      ein[q][r][2 * GD] = (_Float16)fast_sqrtf(ss + so + 1e-12f);  // uniform pair
    }
    __syncthreads();

    // layer 0 for both tiles: 8 independent WMMA chains
    v8f c0[2][4];
    {
      v16h a0 = lds_a_frag(&ein[0][r][0], hsel);
      v16h a1 = lds_a_frag(&ein[1][r][0], hsel);
#pragma unroll
      for (int nt = 0; nt < 4; ++nt) {
        c0[0][nt] = wmma16(a0, w0f[0][nt], vzero);
        c0[1][nt] = wmma16(a1, w0f[0][nt], vzero);
      }
    }
#pragma unroll
    for (int kk = 1; kk < 3; ++kk) {
      v16h a0 = lds_a_frag(&ein[0][r][kk * 32], hsel);
      v16h a1 = lds_a_frag(&ein[1][r][kk * 32], hsel);
#pragma unroll
      for (int nt = 0; nt < 4; ++nt) {
        c0[0][nt] = wmma16(a0, w0f[kk][nt], c0[0][nt]);
        c0[1][nt] = wmma16(a1, w0f[kk][nt], c0[1][nt]);
      }
    }
#pragma unroll
    for (int q = 0; q < 2; ++q)
#pragma unroll
      for (int nt = 0; nt < 4; ++nt) {
        v8f v = leaky8(c0[q][nt]);
#pragma unroll
        for (int e = 0; e < 8; ++e)
          act0[q][e + 8 * hsel][nt * 16 + r] = (_Float16)v[e];
      }
    __syncthreads();

    // layer 1 for both tiles
    v8f c1[2][4];
    {
      v16h a0 = lds_a_frag(&act0[0][r][0], hsel);
      v16h a1 = lds_a_frag(&act0[1][r][0], hsel);
#pragma unroll
      for (int nt = 0; nt < 4; ++nt) {
        c1[0][nt] = wmma16(a0, w1f[0][nt], vzero);
        c1[1][nt] = wmma16(a1, w1f[0][nt], vzero);
      }
    }
    {
      v16h a0 = lds_a_frag(&act0[0][r][32], hsel);
      v16h a1 = lds_a_frag(&act0[1][r][32], hsel);
#pragma unroll
      for (int nt = 0; nt < 4; ++nt) {
        c1[0][nt] = wmma16(a0, w1f[1][nt], c1[0][nt]);
        c1[1][nt] = wmma16(a1, w1f[1][nt], c1[1][nt]);
      }
    }
#pragma unroll
    for (int q = 0; q < 2; ++q)
#pragma unroll
      for (int nt = 0; nt < 4; ++nt)
#pragma unroll
        for (int e = 0; e < 8; ++e) {
          float pre = c1[q][nt][e] + bia1[nt];
          aggc[nt][e] += fmaxf(pre, LALPHA * pre);
        }
    __syncthreads();
  }

  // --- tail tile: j = 96..99 (rows m >= 4 masked, no EXEC divergence) ------
  {
#if HAVE_TDM
    __builtin_amdgcn_s_wait_tensorcnt(0);
    __syncthreads();
#endif
    {
      float hj[16];
#if HAVE_TDM
      const float4* hjp = (const float4*)&hjbuf[2][r][ks];
#else
      const float* hjr = (96 + r < GN) ? (hbase + (size_t)(96 + r) * GD) : hbase;
      const float4* hjp = (const float4*)(hjr + ks);
#endif
#pragma unroll
      for (int qq = 0; qq < 4; ++qq) {
        float4 v = hjp[qq];
        hj[4 * qq + 0] = v.x; hj[4 * qq + 1] = v.y;
        hj[4 * qq + 2] = v.z; hj[4 * qq + 3] = v.w;
      }
      float ss = 0.0f;
#pragma unroll
      for (int e = 0; e < 16; ++e) {
        float d = hik[e] - hj[e];
        ss += d * d;
      }
      v8h q0, q1;
#pragma unroll
      for (int e = 0; e < 8; ++e) {
        q0[e] = (_Float16)hj[e];
        q1[e] = (_Float16)hj[8 + e];
      }
      *(v8h*)&ein[0][r][GD + ks]     = q0;
      *(v8h*)&ein[0][r][GD + ks + 8] = q1;
      float so = __shfl_xor(ss, 16, 32);
      ein[0][r][2 * GD] = (_Float16)fast_sqrtf(ss + so + 1e-12f);
    }
    __syncthreads();

    v8f c0[4];
    {
      v16h a = lds_a_frag(&ein[0][r][0], hsel);
#pragma unroll
      for (int nt = 0; nt < 4; ++nt) c0[nt] = wmma16(a, w0f[0][nt], vzero);
    }
#pragma unroll
    for (int kk = 1; kk < 3; ++kk) {
      v16h a = lds_a_frag(&ein[0][r][kk * 32], hsel);
#pragma unroll
      for (int nt = 0; nt < 4; ++nt) c0[nt] = wmma16(a, w0f[kk][nt], c0[nt]);
    }
#pragma unroll
    for (int nt = 0; nt < 4; ++nt) {
      v8f v = leaky8(c0[nt]);
#pragma unroll
      for (int e = 0; e < 8; ++e)
        act0[0][e + 8 * hsel][nt * 16 + r] = (_Float16)v[e];
    }
    __syncthreads();

    v8f c1[4];
    {
      v16h a = lds_a_frag(&act0[0][r][0], hsel);
#pragma unroll
      for (int nt = 0; nt < 4; ++nt) c1[nt] = wmma16(a, w1f[0][nt], vzero);
    }
    {
      v16h a = lds_a_frag(&act0[0][r][32], hsel);
#pragma unroll
      for (int nt = 0; nt < 4; ++nt) c1[nt] = wmma16(a, w1f[1][nt], c1[nt]);
    }
    // valid tail rows: m = e + 8*hsel < 4  -> e < 4 on lanes with hsel==0
#pragma unroll
    for (int nt = 0; nt < 4; ++nt)
#pragma unroll
      for (int e = 0; e < 4; ++e) {
        float pre = c1[nt][e] + bia1[nt];
        float v = fmaxf(pre, LALPHA * pre);
        aggc[nt][e] += (hsel == 0) ? v : 0.0f;   // cndmask, no EXEC churn
      }
  }

  // reduce over tile-row dim; every lane stores exactly 2 columns
  float s[4];
#pragma unroll
  for (int nt = 0; nt < 4; ++nt) {
    float v = 0.0f;
#pragma unroll
    for (int e = 0; e < 8; ++e) v += aggc[nt][e];
    v += __shfl_xor(v, 16, 32);
    s[nt] = v;
  }
  {
    float v0 = hsel ? s[2] : s[0];
    float v1 = hsel ? s[3] : s[1];
    float* ap = agg + (size_t)bi * 64 + hsel * 32 + r;
    ap[0]  = v0;
    ap[16] = v1;
  }
}

// ---------------------------------------------------------------------------
// Node MLP: [h | agg] (96) -> 64 -> 32
// ---------------------------------------------------------------------------
__global__ __launch_bounds__(32) void node_mp_kernel(
    const float* __restrict__ h,       // [ROWS][32]
    const float* __restrict__ agg,     // [ROWS][64]
    const _Float16* __restrict__ w0t,  // [64][NKPAD]
    const float* __restrict__ b0,
    const _Float16* __restrict__ w1t,  // [32][64]
    const float* __restrict__ b1,
    float* __restrict__ hout)          // [ROWS][32]
{
  __shared__ __align__(16) _Float16 nin[16][NKPAD];
  __shared__ __align__(16) _Float16 act[16][NHID];
  const int lane = threadIdx.x;
  const int r    = lane & 15;
  const int hsel = lane >> 4;
  const int row0 = blockIdx.x * 16;
  const v8f vzero = {};

  {
    const int row = row0 + r;
    const int cs  = hsel * 48;
    for (int c = cs; c < cs + 48; ++c) {
      float v = (c < GD) ? h[(size_t)row * GD + c] : agg[(size_t)row * 64 + (c - GD)];
      nin[r][c] = (_Float16)v;
    }
  }
  __syncthreads();

  v8f c0[4];
  {
    v16h a = lds_a_frag(&nin[r][0], hsel);
#pragma unroll
    for (int nt = 0; nt < 4; ++nt) {
      v16h bw = *(const v16h*)(w0t + (size_t)(nt * 16 + r) * NKPAD + hsel * 16);
      c0[nt] = wmma16(a, bw, vzero);
    }
  }
#pragma unroll
  for (int kk = 1; kk < 3; ++kk) {
    v16h a = lds_a_frag(&nin[r][kk * 32], hsel);
#pragma unroll
    for (int nt = 0; nt < 4; ++nt) {
      v16h bw = *(const v16h*)(w0t + (size_t)(nt * 16 + r) * NKPAD + kk * 32 + hsel * 16);
      c0[nt] = wmma16(a, bw, c0[nt]);
    }
  }
#pragma unroll
  for (int nt = 0; nt < 4; ++nt) {
    float bv = b0[nt * 16 + r];
#pragma unroll
    for (int e = 0; e < 8; ++e) {
      float pre = c0[nt][e] + bv;
      act[e + 8 * hsel][nt * 16 + r] = (_Float16)fmaxf(pre, LALPHA * pre);
    }
  }
  __syncthreads();

  v8f c1[2];
  {
    v16h a = lds_a_frag(&act[r][0], hsel);
#pragma unroll
    for (int nt = 0; nt < 2; ++nt) {
      v16h bw = *(const v16h*)(w1t + (size_t)(nt * 16 + r) * NHID + hsel * 16);
      c1[nt] = wmma16(a, bw, vzero);
    }
  }
  {
    v16h a = lds_a_frag(&act[r][32], hsel);
#pragma unroll
    for (int nt = 0; nt < 2; ++nt) {
      v16h bw = *(const v16h*)(w1t + (size_t)(nt * 16 + r) * NHID + 32 + hsel * 16);
      c1[nt] = wmma16(a, bw, c1[nt]);
    }
  }
#pragma unroll
  for (int nt = 0; nt < 2; ++nt) {
    float bv = b1[nt * 16 + r];
#pragma unroll
    for (int e = 0; e < 8; ++e) {
      float pre = c1[nt][e] + bv;
      hout[(size_t)(row0 + e + 8 * hsel) * NOUT + nt * 16 + r] = fmaxf(pre, LALPHA * pre);
    }
  }
}

// ---------------------------------------------------------------------------
extern "C" void kernel_launch(void* const* d_in, const int* in_sizes, int n_in,
                              void* d_out, int out_size, void* d_ws, size_t ws_size,
                              hipStream_t stream)
{
  (void)in_sizes; (void)n_in; (void)out_size; (void)ws_size;

  const float* x     = (const float*)d_in[0];
  const float* lin_w = (const float*)d_in[1];
  const float* lin_b = (const float*)d_in[2];
  const float* ew0[2] = {(const float*)d_in[3],  (const float*)d_in[11]};
  const float* eb0[2] = {(const float*)d_in[4],  (const float*)d_in[12]};
  const float* ew1[2] = {(const float*)d_in[5],  (const float*)d_in[13]};
  const float* eb1[2] = {(const float*)d_in[6],  (const float*)d_in[14]};
  const float* nw0[2] = {(const float*)d_in[7],  (const float*)d_in[15]};
  const float* nb0[2] = {(const float*)d_in[8],  (const float*)d_in[16]};
  const float* nw1[2] = {(const float*)d_in[9],  (const float*)d_in[17]};
  const float* nb1[2] = {(const float*)d_in[10], (const float*)d_in[18]};
  const float* out_w = (const float*)d_in[19];
  const float* out_b = (const float*)d_in[20];

  char* ws = (char*)d_ws;
  size_t off = 0;
  auto take = [&](size_t bytes) -> char* {
    char* p = ws + off;
    off = (off + bytes + 255) & ~(size_t)255;
    return p;
  };

  float* hA   = (float*)take((size_t)ROWS * GD * sizeof(float));
  float* hB   = (float*)take((size_t)ROWS * GD * sizeof(float));
  float* aggb = (float*)take((size_t)ROWS * 64 * sizeof(float));
  _Float16 *wE0t[2], *wE1t[2], *wN0t[2], *wN1t[2];
  for (int s = 0; s < 2; ++s) {
    wE0t[s] = (_Float16*)take((size_t)64 * EKPAD * sizeof(_Float16));
    wE1t[s] = (_Float16*)take((size_t)64 * 64 * sizeof(_Float16));
    wN0t[s] = (_Float16*)take((size_t)64 * NKPAD * sizeof(_Float16));
    wN1t[s] = (_Float16*)take((size_t)32 * 64 * sizeof(_Float16));
  }

  for (int s = 0; s < 2; ++s) {
    wt_convert_kernel<<<(64 * EKPAD + 255) / 256, 256, 0, stream>>>(ew0[s], eb0[s], wE0t[s], 65, 64, EKPAD, 65);
    wt_convert_kernel<<<(64 * 64    + 255) / 256, 256, 0, stream>>>(ew1[s], eb1[s], wE1t[s], 64, 64, 64, -1);
    wt_convert_kernel<<<(64 * NKPAD + 255) / 256, 256, 0, stream>>>(nw0[s], nb0[s], wN0t[s], 96, 64, NKPAD, -1);
    wt_convert_kernel<<<(32 * 64    + 255) / 256, 256, 0, stream>>>(nw1[s], nb1[s], wN1t[s], 64, 32, 64, -1);
  }

  lin_kernel<<<(GB * GN * GD + 255) / 256, 256, 0, stream>>>(x, lin_w, lin_b, hA);

  // message-passing step 0
  edge_mp_kernel<<<ROWS, 32, 0, stream>>>(hA, wE0t[0], wE1t[0], eb1[0], aggb);
  node_mp_kernel<<<ROWS / 16, 32, 0, stream>>>(hA, aggb, wN0t[0], nb0[0], wN1t[0], nb1[0], hB);
  // message-passing step 1
  edge_mp_kernel<<<ROWS, 32, 0, stream>>>(hB, wE0t[1], wE1t[1], eb1[1], aggb);
  node_mp_kernel<<<ROWS / 16, 32, 0, stream>>>(hB, aggb, wN0t[1], nb0[1], wN1t[1], nb1[1], hA);

  out_kernel<<<(ROWS * 3 + 255) / 256, 256, 0, stream>>>(hA, out_w, out_b, (float*)d_out);
}
